// MLP_1726576855348
// MI455X (gfx1250) — compile-verified, matrix operand order checked by reading, not run on hardware
//
#include <hip/hip_runtime.h>
#include <hip/hip_bf16.h>
#include <math.h>

typedef __attribute__((ext_vector_type(16))) _Float16 v16h;
typedef __attribute__((ext_vector_type(8)))  float    v8f;

#define NPTS   131072
#define NSAMP  6
#define CHUNK  16384
#define MTILES (CHUNK/16)

// ---------------------------------------------------------------------------
// Hash-grid encode: per point, 6 samples x 10 levels x 8 corners gather from
// the embedding table, erf-weighted mean -> 40 features, stored f16 padded
// to K=64 for WMMA. Constants precomputed (PER_LEVEL_SCALE == 2.0 exactly).
// ---------------------------------------------------------------------------
__global__ void encode_kernel(const float* __restrict__ means,
                              const float* __restrict__ stds,
                              const float* __restrict__ emb,
                              _Float16* __restrict__ A0,
                              int row0)
{
    const float    RESF[10]   = {16.f,32.f,64.f,128.f,256.f,512.f,1024.f,2048.f,4096.f,8192.f};
    const unsigned SIZESL[10] = {4920u,35944u,274632u,2097152u,2097152u,2097152u,
                                 2097152u,2097152u,2097152u,2097152u};
    const int      OFFS[10]   = {0,4920,40864,315496,2412648,4509800,
                                 6606952,8704104,10801256,12898408};
    int r = blockIdx.x * blockDim.x + threadIdx.x;
    if (r >= CHUNK) return;
    int p = row0 + r;

    float acc[40];
    #pragma unroll
    for (int i = 0; i < 40; ++i) acc[i] = 0.f;

    for (int s = 0; s < NSAMP; ++s) {
        const float* mp = means + ((size_t)p * NSAMP + s) * 3;
        float x0 = (mp[0] + 1.f) * 0.5f;
        float x1 = (mp[1] + 1.f) * 0.5f;
        float x2 = (mp[2] + 1.f) * 0.5f;
        float sd = stds[(size_t)p * NSAMP + s];
        #pragma unroll
        for (int l = 0; l < 10; ++l) {
            float res = RESF[l];
            float wgt = erff(1.f / sqrtf(8.f * sd * sd * res * res));
            float p0 = x0 * res + 0.5f, p1 = x1 * res + 0.5f, p2 = x2 * res + 0.5f;
            float f0 = floorf(p0), f1 = floorf(p1), f2 = floorf(p2);
            float fr0 = p0 - f0, fr1 = p1 - f1, fr2 = p2 - f2;
            unsigned b0 = (unsigned)(int)f0, b1 = (unsigned)(int)f1, b2 = (unsigned)(int)f2;
            float g0 = 0.f, g1 = 0.f, g2 = 0.f, g3 = 0.f;
            #pragma unroll
            for (int c = 0; c < 8; ++c) {
                unsigned c0 = c & 1u, c1 = (c >> 1) & 1u, c2 = (c >> 2) & 1u;
                unsigned h = ((b0 + c0) * 1u) ^ ((b1 + c1) * 2654435761u)
                           ^ ((b2 + c2) * 805459861u);
                unsigned idx = h % SIZESL[l];
                float w = (c0 ? fr0 : 1.f - fr0) * (c1 ? fr1 : 1.f - fr1)
                        * (c2 ? fr2 : 1.f - fr2);
                const float* e = emb + ((size_t)(unsigned)OFFS[l] + idx) * 4;
                g0 += w * e[0]; g1 += w * e[1]; g2 += w * e[2]; g3 += w * e[3];
            }
            acc[l*4+0] += wgt * g0; acc[l*4+1] += wgt * g1;
            acc[l*4+2] += wgt * g2; acc[l*4+3] += wgt * g3;
        }
    }
    _Float16* dst = A0 + (size_t)r * 64;
    const float inv = 1.f / (float)NSAMP;
    #pragma unroll
    for (int i = 0; i < 40; ++i) dst[i] = (_Float16)(acc[i] * inv);
    #pragma unroll
    for (int i = 40; i < 64; ++i) dst[i] = (_Float16)0.f;
}

// ---------------------------------------------------------------------------
// View-direction positional encoding: 27 values -> X[256..282] (pad to 320)
// and duplicated into HC[512..538] (pad to 576) for the concat-skip layer.
// ---------------------------------------------------------------------------
__global__ void direnc_kernel(const float* __restrict__ viewdirs,
                              _Float16* __restrict__ X,
                              _Float16* __restrict__ HC,
                              int row0)
{
    int r = blockIdx.x * blockDim.x + threadIdx.x;
    if (r >= CHUNK) return;
    int p = row0 + r;
    float v[3] = { viewdirs[(size_t)p*3+0], viewdirs[(size_t)p*3+1], viewdirs[(size_t)p*3+2] };
    float e[27];
    e[0] = v[0]; e[1] = v[1]; e[2] = v[2];
    int k = 3;
    #pragma unroll
    for (int i = 0; i < 4; ++i) {
        float sc = (float)(1 << i);
        #pragma unroll
        for (int d = 0; d < 3; ++d) e[k++] = sinf(v[d] * sc);
    }
    #pragma unroll
    for (int i = 0; i < 4; ++i) {
        float sc = (float)(1 << i);
        #pragma unroll
        for (int d = 0; d < 3; ++d) e[k++] = cosf(v[d] * sc);   // sin(x + pi/2)
    }
    _Float16* xd = X  + (size_t)r * 320 + 256;
    _Float16* hd = HC + (size_t)r * 576 + 512;
    #pragma unroll
    for (int i = 0; i < 27; ++i) xd[i] = (_Float16)e[i];
    #pragma unroll
    for (int i = 27; i < 64; ++i) xd[i] = (_Float16)0.f;        // cols 283..319
    #pragma unroll
    for (int i = 0; i < 27; ++i) hd[i] = (_Float16)e[i];
    #pragma unroll
    for (int i = 27; i < 64; ++i) hd[i] = (_Float16)0.f;        // cols 539..575
}

// ---------------------------------------------------------------------------
// Pack f32 weights [Kin][Nout] into WMMA B-fragment-major f16 layout:
// frag(kt,nt) = 32 lanes x 16 halves; lane l: n = nt*16 + (l&15), g = l>>4,
// halves v*2+j hold W[kt*32 + ((v<4)?0:16) + (v&3)*2 + g*8 + j][n], zero-pad.
// ---------------------------------------------------------------------------
__global__ void pack_kernel(const float* __restrict__ W, _Float16* __restrict__ out,
                            int Kin, int Nout, int Ktiles, int Ntiles)
{
    int t = blockIdx.x * blockDim.x + threadIdx.x;
    int total = Ktiles * Ntiles * 32;
    if (t >= total) return;
    int lane = t & 31;
    int tile = t >> 5;
    int nt = tile % Ntiles;
    int kt = tile / Ntiles;
    int g = lane >> 4;
    int n = nt * 16 + (lane & 15);
    _Float16* o = out + (size_t)t * 16;
    #pragma unroll
    for (int v = 0; v < 8; ++v) {
        int kbase = kt * 32 + ((v < 4) ? 0 : 16) + (v & 3) * 2 + g * 8;
        #pragma unroll
        for (int j = 0; j < 2; ++j) {
            int kk = kbase + j;
            float val = (kk < Kin && n < Nout) ? W[(size_t)kk * Nout + n] : 0.f;
            o[v*2 + j] = (_Float16)val;
        }
    }
}

// ---------------------------------------------------------------------------
// WMMA GEMM, fully compile-time shaped. Each wave: 16 rows x (NT*16) cols.
// K loop fully unrolled (immediate-offset b128 loads, load-all-then-wmma-all
// per K-tile so the NT wmmas issue back-to-back and next-tile loads overlap).
//   EPI 0: relu -> f16 out
//   EPI 1: identity -> f16 out AND dup into out2 (concat copy);
//          col 0 -> softplus density into fout[NPTS*3 + global_row]
//   EPI 2: sigmoid rgb (cols < nreal) -> fout[global_row*3 + col]
// ---------------------------------------------------------------------------
template<int NT, int EPI, int KT, int NTILES, int KPAD>
__global__ void __launch_bounds__(256)
gemm_wmma(const _Float16* __restrict__ A,
          const _Float16* __restrict__ Bp,
          const float* __restrict__ bias, int nreal,
          _Float16* __restrict__ out, int ostride,
          _Float16* __restrict__ out2, int o2stride, int o2coloff,
          float* __restrict__ fout, int row0)
{
    constexpr int NGRP = NTILES / NT;
    int lane = threadIdx.x & 31;
    int wid  = (blockIdx.x * blockDim.x + threadIdx.x) >> 5;
    int tm = wid / NGRP;
    int tn = wid % NGRP;
    if (tm >= MTILES) return;
    int g = lane >> 4;
    int m = lane & 15;

    const _Float16* arow = A + (size_t)(tm * 16 + m) * KPAD;
    const v16h* bbase = (const v16h*)Bp + (size_t)(tn * NT) * 32 + lane;

    v8f acc[NT];
    #pragma unroll
    for (int j = 0; j < NT; ++j)
        #pragma unroll
        for (int e = 0; e < 8; ++e) acc[j][e] = 0.f;

    #pragma unroll
    for (int kt = 0; kt < KT; ++kt) {
        // A fragment (ISA 16-bit A 16x32 per-lane K-pair layout), immediate offsets.
        union { v16h h; unsigned u[8]; } a;
        #pragma unroll
        for (int v = 0; v < 8; ++v) {
            int kb = kt * 32 + ((v < 4) ? 0 : 16) + (v & 3) * 2 + g * 8;
            a.u[v] = *(const unsigned*)(arow + kb);
        }
        // All NT B fragments for this K-tile, immediate offsets.
        v16h bf[NT];
        #pragma unroll
        for (int j = 0; j < NT; ++j)
            bf[j] = bbase[(size_t)kt * NTILES * 32 + j * 32];
        // NT independent WMMAs (shared A, distinct B/acc) issue back-to-back.
        #pragma unroll
        for (int j = 0; j < NT; ++j)
            acc[j] = __builtin_amdgcn_wmma_f32_16x16x32_f16(
                         false, a.h, false, bf[j], (short)0, acc[j], false, false);
    }

    // Epilogue. C/D layout: VGPR v, lanes 0-15: M=v, lanes 16-31: M=v+8.
    #pragma unroll
    for (int j = 0; j < NT; ++j) {
        int ncol = tn * (NT * 16) + j * 16 + m;
        float bs = (ncol < nreal) ? bias[ncol] : 0.f;
        #pragma unroll
        for (int v = 0; v < 8; ++v) {
            int row = tm * 16 + v + g * 8;
            float val = acc[j][v] + bs;
            if (EPI == 0) {
                val = val > 0.f ? val : 0.f;
                out[(size_t)row * ostride + ncol] = (_Float16)val;
            } else if (EPI == 1) {
                out[(size_t)row * ostride + ncol] = (_Float16)val;
                out2[(size_t)row * o2stride + o2coloff + ncol] = (_Float16)val;
                if (ncol == 0) {
                    float x = val - 1.0f;   // DENSITY_BIAS
                    float dens = x > 0.f ? x + log1pf(expf(-x)) : log1pf(expf(x));
                    fout[(size_t)NPTS * 3 + row0 + row] = dens;
                }
            } else { // EPI == 2: rgb
                if (ncol < nreal) {
                    float sg = 1.f / (1.f + expf(-val));
                    fout[(size_t)(row0 + row) * 3 + ncol] = sg * 1.002f - 0.001f;
                }
            }
        }
    }
}

// ---------------------------------------------------------------------------
extern "C" void kernel_launch(void* const* d_in, const int* in_sizes, int n_in,
                              void* d_out, int out_size, void* d_ws, size_t ws_size,
                              hipStream_t stream)
{
    const float* means    = (const float*)d_in[1];
    const float* stds     = (const float*)d_in[2];
    const float* viewdirs = (const float*)d_in[3];
    const float* emb      = (const float*)d_in[5];
    const float* W1   = (const float*)d_in[6];
    const float* b1   = (const float*)d_in[7];
    const float* W2   = (const float*)d_in[8];
    const float* b2   = (const float*)d_in[9];
    const float* Wv0  = (const float*)d_in[10];
    const float* bv0  = (const float*)d_in[11];
    const float* Wv1  = (const float*)d_in[12];
    const float* bv1  = (const float*)d_in[13];
    const float* Wrgb = (const float*)d_in[14];
    const float* brgb = (const float*)d_in[15];
    float* out = (float*)d_out;

    // Workspace arena (256B aligned slices)
    auto al = [](size_t x) { return (x + 255) & ~(size_t)255; };
    size_t szW1p  = al((size_t)2  * 4  * 1024);   // Ktiles*Ntiles*32*16*2B
    size_t szW2p  = al((size_t)2  * 16 * 1024);
    size_t szWv0p = al((size_t)10 * 16 * 1024);
    size_t szWv1p = al((size_t)18 * 16 * 1024);
    size_t szWrp  = al((size_t)8  * 1  * 1024);
    size_t szA0   = al((size_t)CHUNK * 64  * 2);
    size_t szH1   = al((size_t)CHUNK * 64  * 2);
    size_t szX    = al((size_t)CHUNK * 320 * 2);
    size_t szHC   = al((size_t)CHUNK * 576 * 2);
    size_t szH1b  = al((size_t)CHUNK * 256 * 2);
    size_t need = szW1p + szW2p + szWv0p + szWv1p + szWrp
                + szA0 + szH1 + szX + szHC + szH1b;
    if (ws_size < need) return;

    char* p = (char*)d_ws;
    _Float16* W1p  = (_Float16*)p; p += szW1p;
    _Float16* W2p  = (_Float16*)p; p += szW2p;
    _Float16* Wv0p = (_Float16*)p; p += szWv0p;
    _Float16* Wv1p = (_Float16*)p; p += szWv1p;
    _Float16* Wrp  = (_Float16*)p; p += szWrp;
    _Float16* A0   = (_Float16*)p; p += szA0;
    _Float16* H1   = (_Float16*)p; p += szH1;
    _Float16* X    = (_Float16*)p; p += szX;
    _Float16* HC   = (_Float16*)p; p += szHC;
    _Float16* H1b  = (_Float16*)p; p += szH1b;

    // Pack weights once (fragment-major f16, zero padded in K and N).
    pack_kernel<<<(2*4*32   + 255) / 256, 256, 0, stream>>>(W1,   W1p,  40,  64,  2,  4);
    pack_kernel<<<(2*16*32  + 255) / 256, 256, 0, stream>>>(W2,   W2p,  64,  256, 2,  16);
    pack_kernel<<<(10*16*32 + 255) / 256, 256, 0, stream>>>(Wv0,  Wv0p, 283, 256, 10, 16);
    pack_kernel<<<(18*16*32 + 255) / 256, 256, 0, stream>>>(Wv1,  Wv1p, 539, 256, 18, 16);
    pack_kernel<<<(8*1*32   + 255) / 256, 256, 0, stream>>>(Wrgb, Wrp,  256, 3,   8,  1);

    for (int row0 = 0; row0 < NPTS; row0 += CHUNK) {
        encode_kernel<<<CHUNK / 256, 256, 0, stream>>>(means, stds, emb, A0, row0);
        direnc_kernel<<<CHUNK / 256, 256, 0, stream>>>(viewdirs, X, HC, row0);

        // L1: features(40->64) relu
        gemm_wmma<4, 0, 2, 4, 64><<<MTILES / 8, 256, 0, stream>>>(
            A0, W1p, b1, 64, H1, 64, nullptr, 0, 0, nullptr, row0);
        // L2: 64->256, density + dup x into HC[256..511]
        gemm_wmma<4, 1, 2, 16, 64><<<MTILES * 4 / 8, 256, 0, stream>>>(
            H1, W2p, b2, 256, X, 320, HC, 576, 256, out, row0);
        // L3: inp(283->256, Kpad 320) relu -> HC[0..255]
        gemm_wmma<4, 0, 10, 16, 320><<<MTILES * 4 / 8, 256, 0, stream>>>(
            X, Wv0p, bv0, 256, HC, 576, nullptr, 0, 0, nullptr, row0);
        // L4: concat(539->256, Kpad 576) relu
        gemm_wmma<4, 0, 18, 16, 576><<<MTILES * 4 / 8, 256, 0, stream>>>(
            HC, Wv1p, bv1, 256, H1b, 256, nullptr, 0, 0, nullptr, row0);
        // L5: 256->3 sigmoid rgb
        gemm_wmma<1, 2, 8, 1, 256><<<MTILES / 8, 256, 0, stream>>>(
            H1b, Wrp, brgb, 3, nullptr, 0, nullptr, 0, 0, out, row0);
    }
}
